// Model_15212774162686
// MI455X (gfx1250) — matrix operand order repaired and checked by reading, not compiled, not run on hardware
//
#include <hip/hip_runtime.h>
#include <hip/hip_bf16.h>
#include <math.h>

// ---------------------------------------------------------------------------
// MI455X (gfx1250) implementation.
// GEMMs run on the bf16 WMMA pipe using a 3-term hi/lo split of f32 operands
// (hi*hi + hi*lo + lo*hi, f32 accumulation) -> ~f32 accuracy at ~2.7x the
// native f32-WMMA rate.
// Tile: workgroup 64x128x32, 8 wave32 (2x4), wave tile 32x32 -> 4 accumulators,
// 12 v_wmma per K-block vs 16 ds_load_b128.  B staging pairs adjacent k-rows
// so every transposed LDS write is a packed 32-bit (k,k+1) store (halves the
// DS-store instruction count vs 16-bit scatter).  Next-K-block
// global_prefetch_b8 overlaps L2->WGP$ fill with WMMA compute.
// ---------------------------------------------------------------------------

typedef __attribute__((ext_vector_type(16))) __bf16 v16bf;
typedef __attribute__((ext_vector_type(8)))  float  v8f;

#define LDP 40  // LDS row pitch in ushorts: 32 data + 8 pad (80B, 16B aligned)

__device__ __forceinline__ unsigned short f32_to_bf16_rn(float x) {
  unsigned int u = __float_as_uint(x);
  u += 0x7FFFu + ((u >> 16) & 1u);
  return (unsigned short)(u >> 16);
}

__device__ __forceinline__ void bf_split(float x, unsigned short& hi, unsigned short& lo) {
  unsigned int u = __float_as_uint(x);
  hi = (unsigned short)(u >> 16);                      // truncated hi (exact bf16)
  float hf = __uint_as_float(u & 0xFFFF0000u);
  lo = f32_to_bf16_rn(x - hf);                         // residual, RNE
}

// pack two bf16 (k even in [15:0], k+1 in [31:16]) — matches 16-bit B layout
__device__ __forceinline__ unsigned int pack2(unsigned short a, unsigned short b) {
  return (unsigned int)a | ((unsigned int)b << 16);
}

union FragBF { uint4 q[2]; v16bf v; };

#define WMMA_BF16(A_, B_, C_) \
  __builtin_amdgcn_wmma_f32_16x16x32_bf16(false, (A_), false, (B_), (short)0, (C_), false, false)

// C[M,N] = op(A[M,K] @ B[K,N] (+ bias) (relu?)), row-major, M%64==0, N%128==0, K%32==0
__global__ __launch_bounds__(256)
void gemm_bf16x3(const float* __restrict__ A, int lda,
                 const float* __restrict__ B, int ldb,
                 float* __restrict__ C, int ldc,
                 int K, const float* __restrict__ bias, int relu)
{
  __shared__ uint4 smem_raw[1920];                     // 30 KB
  unsigned short* As_hi = (unsigned short*)smem_raw;   // [64][LDP]   (m,k)
  unsigned short* As_lo = As_hi + 64 * LDP;
  unsigned short* Bs_hi = As_lo + 64 * LDP;            // [128][LDP]  (n,k) transposed
  unsigned short* Bs_lo = Bs_hi + 128 * LDP;

  const int tid  = threadIdx.x;
  const int lane = tid & 31;
  const int wv   = tid >> 5;
  const int m0   = blockIdx.y * 64;
  const int n0   = blockIdx.x * 128;

  // wave tile: 32x32 -> 2x2 subtiles of 16x16
  const int r0   = (wv & 1) * 32;          // wave row base within 64
  const int c0   = (wv >> 1) * 32;         // wave col base within 128
  const int ka   = (lane & 16) ? 16 : 0;   // A frag K-half (contiguous k, matches B)
  const int kb0  = (lane & 16) ? 16 : 0;   // B frag K-half (contiguous k, matches A)
  const int l15  = lane & 15;

  v8f acc00 = {}, acc01 = {}, acc10 = {}, acc11 = {};

  for (int kb = 0; kb < K; kb += 32) {
    const int kbn = (kb + 32 < K) ? (kb + 32) : kb;    // next-block prefetch base
    __syncthreads();
    // ---- stage A tile 64x32 (f32 -> hi/lo bf16, row-major in LDS) ----
#pragma unroll
    for (int i = 0; i < 2; ++i) {
      int id  = tid + i * 256;             // 0..511 float4s
      int row = id >> 3;
      int c4  = (id & 7) << 2;
      const float4 a = *(const float4*)(A + (size_t)(m0 + row) * lda + kb + c4);
      __builtin_prefetch(A + (size_t)(m0 + row) * lda + kbn + c4, 0, 3);
      unsigned short h0,h1,h2,h3,l0,l1,l2,l3;
      bf_split(a.x,h0,l0); bf_split(a.y,h1,l1); bf_split(a.z,h2,l2); bf_split(a.w,h3,l3);
      unsigned int* dh = (unsigned int*)&As_hi[row * LDP + c4];
      unsigned int* dl = (unsigned int*)&As_lo[row * LDP + c4];
      dh[0] = pack2(h0, h1);  dh[1] = pack2(h2, h3);
      dl[0] = pack2(l0, l1);  dl[1] = pack2(l2, l3);
    }
    // ---- stage B tile 32x128 transposed -> LDS[n][k], k paired so every
    //      store is a packed 32-bit (k,k+1) dword ----
#pragma unroll
    for (int i = 0; i < 2; ++i) {
      int id = tid + i * 256;              // 0..511 work items
      int kr = (id >> 5) << 1;             // even k row 0..30
      int n4 = (id & 31) << 2;             // 0..124
      const float4 b0 = *(const float4*)(B + (size_t)(kb + kr)     * ldb + n0 + n4);
      const float4 b1 = *(const float4*)(B + (size_t)(kb + kr + 1) * ldb + n0 + n4);
      __builtin_prefetch(B + (size_t)(kbn + kr)     * ldb + n0 + n4, 0, 3);
      __builtin_prefetch(B + (size_t)(kbn + kr + 1) * ldb + n0 + n4, 0, 3);
      unsigned short ha, la, hb, lb;
      bf_split(b0.x, ha, la); bf_split(b1.x, hb, lb);
      *(unsigned int*)&Bs_hi[(n4+0)*LDP + kr] = pack2(ha, hb);
      *(unsigned int*)&Bs_lo[(n4+0)*LDP + kr] = pack2(la, lb);
      bf_split(b0.y, ha, la); bf_split(b1.y, hb, lb);
      *(unsigned int*)&Bs_hi[(n4+1)*LDP + kr] = pack2(ha, hb);
      *(unsigned int*)&Bs_lo[(n4+1)*LDP + kr] = pack2(la, lb);
      bf_split(b0.z, ha, la); bf_split(b1.z, hb, lb);
      *(unsigned int*)&Bs_hi[(n4+2)*LDP + kr] = pack2(ha, hb);
      *(unsigned int*)&Bs_lo[(n4+2)*LDP + kr] = pack2(la, lb);
      bf_split(b0.w, ha, la); bf_split(b1.w, hb, lb);
      *(unsigned int*)&Bs_hi[(n4+3)*LDP + kr] = pack2(ha, hb);
      *(unsigned int*)&Bs_lo[(n4+3)*LDP + kr] = pack2(la, lb);
    }
    __syncthreads();

    // ---- fragment loads (contiguous b128 LDS reads, k = 16*half + s for A and B) ----
    FragBF a0h, a0l, a1h, a1l, b0h, b0l, b1h, b1l;
    {
      const int mr0 = r0 + l15, mr1 = r0 + 16 + l15;
      const uint4* p;
      p = (const uint4*)&As_hi[mr0 * LDP + ka]; a0h.q[0] = p[0]; a0h.q[1] = p[1];
      p = (const uint4*)&As_lo[mr0 * LDP + ka]; a0l.q[0] = p[0]; a0l.q[1] = p[1];
      p = (const uint4*)&As_hi[mr1 * LDP + ka]; a1h.q[0] = p[0]; a1h.q[1] = p[1];
      p = (const uint4*)&As_lo[mr1 * LDP + ka]; a1l.q[0] = p[0]; a1l.q[1] = p[1];
      const int nc0 = c0 + l15, nc1 = c0 + 16 + l15;
      p = (const uint4*)&Bs_hi[nc0 * LDP + kb0]; b0h.q[0] = p[0]; b0h.q[1] = p[1];
      p = (const uint4*)&Bs_lo[nc0 * LDP + kb0]; b0l.q[0] = p[0]; b0l.q[1] = p[1];
      p = (const uint4*)&Bs_hi[nc1 * LDP + kb0]; b1h.q[0] = p[0]; b1h.q[1] = p[1];
      p = (const uint4*)&Bs_lo[nc1 * LDP + kb0]; b1l.q[0] = p[0]; b1l.q[1] = p[1];
    }

    // ---- 12 WMMAs; 4 independent accumulators interleaved to hide RAW ----
    acc00 = WMMA_BF16(a0h.v, b0h.v, acc00);
    acc01 = WMMA_BF16(a0h.v, b1h.v, acc01);
    acc10 = WMMA_BF16(a1h.v, b0h.v, acc10);
    acc11 = WMMA_BF16(a1h.v, b1h.v, acc11);
    acc00 = WMMA_BF16(a0h.v, b0l.v, acc00);
    acc01 = WMMA_BF16(a0h.v, b1l.v, acc01);
    acc10 = WMMA_BF16(a1h.v, b0l.v, acc10);
    acc11 = WMMA_BF16(a1h.v, b1l.v, acc11);
    acc00 = WMMA_BF16(a0l.v, b0h.v, acc00);
    acc01 = WMMA_BF16(a0l.v, b1h.v, acc01);
    acc10 = WMMA_BF16(a1l.v, b0h.v, acc10);
    acc11 = WMMA_BF16(a1l.v, b1h.v, acc11);
  }

  // ---- epilogue: C/D layout = VGPR j -> M=j (lanes 0-15) / M=j+8 (16-31) ----
  const int rhalf = (lane & 16) ? 8 : 0;
  const int col0  = n0 + c0 + l15;
  const int col1  = col0 + 16;
  const float bi0 = bias ? bias[col0] : 0.f;
  const float bi1 = bias ? bias[col1] : 0.f;
#pragma unroll
  for (int ms = 0; ms < 2; ++ms) {
    const int crow = m0 + r0 + ms * 16 + rhalf;
    const v8f& a0 = ms ? acc10 : acc00;
    const v8f& a1 = ms ? acc11 : acc01;
#pragma unroll
    for (int j = 0; j < 8; ++j) {
      float v0 = a0[j] + bi0;
      float v1 = a1[j] + bi1;
      if (relu) { v0 = fmaxf(v0, 0.f); v1 = fmaxf(v1, 0.f); }
      C[(size_t)(crow + j) * ldc + col0] = v0;
      C[(size_t)(crow + j) * ldc + col1] = v1;
    }
  }
}

// ---------------------------------------------------------------------------
// Edge scoring: one wave32 per edge; D=128 -> float4 per lane.
// ---------------------------------------------------------------------------
__device__ __forceinline__ float wave_sum32(float v) {
#pragma unroll
  for (int o = 16; o > 0; o >>= 1) v += __shfl_xor(v, o, 32);
  return v;
}
__device__ __forceinline__ float dot4w(float4 a, float4 b, float4 w) {
  return a.x*b.x*w.x + a.y*b.y*w.y + a.z*b.z*w.z + a.w*b.w*w.w;
}
__device__ __forceinline__ float sigmoidf_(float x) { return 1.f / (1.f + __expf(-x)); }

__global__ __launch_bounds__(256)
void edge_score_kernel(const int* __restrict__ edges,
                       const float* __restrict__ znm,   // z_node_mean [2048,128]
                       const float* __restrict__ zem,   // z_edge_mean [3072,128]
                       const float* __restrict__ mir1, const float* __restrict__ mir2,
                       const float* __restrict__ mir3,
                       const float* __restrict__ dis1, const float* __restrict__ dis2,
                       const float* __restrict__ dis3,
                       const float* __restrict__ W_li, const float* __restrict__ b_li,
                       float* __restrict__ out, int E)
{
  const int e    = blockIdx.x * 8 + (threadIdx.x >> 5);
  const int lane = threadIdx.x & 31;
  if (e >= E) return;
  const int m = edges[2 * e + 0];
  const int d = edges[2 * e + 1];
  const float4 w  = ((const float4*)W_li)[lane];
  const float  bl = b_li[0];

  const float4 zn = ((const float4*)(znm  + (size_t)d * 128))[lane];
  const float4 ze = ((const float4*)(zem  + (size_t)m * 128))[lane];
  const float4 m1 = ((const float4*)(mir1 + (size_t)m * 128))[lane];
  const float4 m2 = ((const float4*)(mir2 + (size_t)m * 128))[lane];
  const float4 m3 = ((const float4*)(mir3 + (size_t)m * 128))[lane];
  const float4 d1 = ((const float4*)(dis1 + (size_t)d * 128))[lane];
  const float4 d2 = ((const float4*)(dis2 + (size_t)d * 128))[lane];
  const float4 d3 = ((const float4*)(dis3 + (size_t)d * 128))[lane];

  float recon = wave_sum32(zn.x*ze.x + zn.y*ze.y + zn.z*ze.z + zn.w*ze.w);
  float pauto = sigmoidf_(wave_sum32(dot4w(ze, zn, w)) + bl);
  float pf1   = sigmoidf_(wave_sum32(dot4w(m1, d1, w)) + bl);
  float pf2   = sigmoidf_(wave_sum32(dot4w(m2, d2, w)) + bl);
  float pf3   = sigmoidf_(wave_sum32(dot4w(m3, d3, w)) + bl);
  float si    = (pf1 + pf2 + pf3) * (1.f / 3.f);
  float con   = 0.1f * pauto + 0.9f * si;

  if (lane == 0) {
    out[0 * (size_t)E + e] = recon;
    out[1 * (size_t)E + e] = pauto;
    out[2 * (size_t)E + e] = pf1;
    out[3 * (size_t)E + e] = pf2;
    out[4 * (size_t)E + e] = pf3;
    out[5 * (size_t)E + e] = con;
    out[6 * (size_t)E + e] = si;
  }
}

// ---------------------------------------------------------------------------
static inline void launch_gemm(const float* A, const float* B, float* C,
                               int M, int N, int K, const float* bias, int relu,
                               hipStream_t s) {
  dim3 grid(N / 128, M / 64), block(256);
  gemm_bf16x3<<<grid, block, 0, s>>>(A, K, B, N, C, N, K, bias, relu);
}

extern "C" void kernel_launch(void* const* d_in, const int* in_sizes, int n_in,
                              void* d_out, int out_size, void* d_ws, size_t ws_size,
                              hipStream_t stream) {
  const int N_D = 2048, N_M = 3072, H1 = 512, D = 128;
  const int E = in_sizes[10] / 2;

  const float* AT       = (const float*)d_in[0];
  const float* Ain      = (const float*)d_in[1];
  const float* HMG      = (const float*)d_in[2];
  const float* HDG      = (const float*)d_in[3];
  const float* mir_feat = (const float*)d_in[4];
  const float* dis_feat = (const float*)d_in[5];
  const float* HMD      = (const float*)d_in[6];
  const float* HDM      = (const float*)d_in[7];
  const float* HMM      = (const float*)d_in[8];
  const float* HDD      = (const float*)d_in[9];
  const int*   edges    = (const int*)d_in[10];
  const float* W_ne     = (const float*)d_in[11];
  const float* b_ne     = (const float*)d_in[12];
  const float* W_he     = (const float*)d_in[13];
  const float* b_he     = (const float*)d_in[14];
  const float* W_mun    = (const float*)d_in[15];
  const float* b_mun    = (const float*)d_in[16];
  const float* W_mue    = (const float*)d_in[17];
  const float* b_mue    = (const float*)d_in[18];
  const float* W1m      = (const float*)d_in[19];
  const float* W2m      = (const float*)d_in[20];
  const float* W1d      = (const float*)d_in[21];
  const float* W2d      = (const float*)d_in[22];
  const float* W_li     = (const float*)d_in[23];
  const float* b_li     = (const float*)d_in[24];

  // workspace carve (floats)
  float* w = (float*)d_ws;
  float* z_node      = w; w += (size_t)N_D * H1;
  float* z_edge      = w; w += (size_t)N_M * H1;
  float* z_node_mean = w; w += (size_t)N_D * D;
  float* z_edge_mean = w; w += (size_t)N_M * D;
  float* P_m         = w; w += (size_t)N_M * N_M;   // mir_feat @ W1m (hoisted, shared)
  float* P_d         = w; w += (size_t)N_D * N_D;   // dis_feat @ W1d (hoisted, shared)
  float* hbuf        = w; w += (size_t)N_M * N_M;   // relu(G @ P), reused
  float* tbuf        = w; w += (size_t)N_M * D;     // h @ W2, reused

  float* out = (float*)d_out;
  float* mir_out[3] = { out + 7*(size_t)E,
                        out + 7*(size_t)E + 1*(size_t)N_M*D,
                        out + 7*(size_t)E + 2*(size_t)N_M*D };
  float* dis_out[3] = { out + 7*(size_t)E + 3*(size_t)N_M*D,
                        out + 7*(size_t)E + 3*(size_t)N_M*D + 1*(size_t)N_D*D,
                        out + 7*(size_t)E + 3*(size_t)N_M*D + 2*(size_t)N_D*D };
  const float* mirG[3] = { HMG, HMM, HMD };   // mir1, mir2, mir3
  const float* disG[3] = { HDG, HDM, HDD };   // dis1, dis2, dis3

  // encoder path
  launch_gemm(AT,     W_ne,  z_node,      N_D, H1, N_M, b_ne,  1, stream);
  launch_gemm(Ain,    W_he,  z_edge,      N_M, H1, N_D, b_he,  1, stream);
  launch_gemm(z_node, W_mun, z_node_mean, N_D, D,  H1,  b_mun, 0, stream);
  launch_gemm(z_edge, W_mue, z_edge_mean, N_M, D,  H1,  b_mue, 0, stream);

  // hoisted shared products
  launch_gemm(mir_feat, W1m, P_m, N_M, N_M, N_M, nullptr, 0, stream);
  launch_gemm(dis_feat, W1d, P_d, N_D, N_D, N_D, nullptr, 0, stream);

  // three hypergraph passes (hbuf/tbuf reused; same-stream ordering)
  for (int i = 0; i < 3; ++i) {
    launch_gemm(mirG[i], P_m,  hbuf,       N_M, N_M, N_M, nullptr, 1, stream);
    launch_gemm(hbuf,    W2m,  tbuf,       N_M, D,   N_M, nullptr, 0, stream);
    launch_gemm(mirG[i], tbuf, mir_out[i], N_M, D,   N_M, nullptr, 0, stream);
  }
  for (int i = 0; i < 3; ++i) {
    launch_gemm(disG[i], P_d,  hbuf,       N_D, N_D, N_D, nullptr, 1, stream);
    launch_gemm(hbuf,    W2d,  tbuf,       N_D, D,   N_D, nullptr, 0, stream);
    launch_gemm(disG[i], tbuf, dis_out[i], N_D, D,   N_D, nullptr, 0, stream);
  }

  // fused edge gather + scoring (7 per-edge outputs)
  dim3 eg((E + 7) / 8), eb(256);
  edge_score_kernel<<<eg, eb, 0, stream>>>(edges, z_node_mean, z_edge_mean,
                                           mir_out[0], mir_out[1], mir_out[2],
                                           dis_out[0], dis_out[1], dis_out[2],
                                           W_li, b_li, out, E);
}
